// FrontierExplorationPolicy_17935783428766
// MI455X (gfx1250) — matrix operand order repaired and checked by reading, not compiled, not run on hardware
//
#include <hip/hip_runtime.h>
#include <hip/hip_bf16.h>

typedef __attribute__((ext_vector_type(16))) _Float16 v16h;
typedef __attribute__((ext_vector_type(8)))  _Float16 v8h;
typedef __attribute__((ext_vector_type(8)))  float    v8f;

#define MDIM 1024
#define NBATCH 16
#define NCH 8
#define EXPLORED_MAP 1

// ---------------------------------------------------------------------------
// Precomputed B-matrix table for the 21x21 disk conv expressed as GEMM.
// For kernel row ky, output tile column n, K = input-column index:
//   B1[K][n]  = w(ky, K-n)       for K  in [0,32)
//   B2[K'][n] = w(ky, 32+K'-n)   for K' in [0,32) (only K'<4 can be nonzero;
//                                all other entries are stored as zero)
// Stored per-lane in this kernel's wave32 WMMA B-operand layout (contiguous k, same as A):
//   lane l holds column n = l&15, K-half base kb = (l<16)?0:16,
//   vector element e <-> K = kb+e.
// Layout: [ky][lane][32] f16 bit patterns (0x3C00 = 1.0h); full 16-element
// storage for both operands so the kernel feeds WMMA with pure b128 loads
// (no register-assembly movs / zero fills).
// ---------------------------------------------------------------------------
struct BTab { unsigned short v[21 * 32 * 32]; };

static constexpr bool in_disk(int kx, int ky) {
  const int dx = kx - 10, dy = ky - 10;
  return kx >= 0 && kx <= 20 && (dx * dx + dy * dy) <= 100;
}

static constexpr BTab make_btab() {
  BTab t{};
  for (int ky = 0; ky < 21; ++ky)
    for (int lane = 0; lane < 32; ++lane) {
      const int n  = lane & 15;
      const int kb = (lane < 16) ? 0 : 16;
      for (int e = 0; e < 16; ++e)
        t.v[(ky * 32 + lane) * 32 + e] = in_disk(kb + e - n, ky) ? 0x3C00 : 0;
      for (int e = 0; e < 16; ++e)
        t.v[(ky * 32 + lane) * 32 + 16 + e] =
            in_disk(32 + kb + e - n, ky) ? 0x3C00 : 0;
    }
  return t;
}

__constant__ BTab g_btab = make_btab();

// ---------------------------------------------------------------------------
// Kernel 1: binarize + 21x21 disk erosion via WMMA conv, -> eroded mask (f16).
// Block = 512 threads = 16 wave32 waves; each wave owns one 16x16 output tile
// of a 64x64 block region. Halo (88x88 f16, incl. slack so the zero-weighted
// K'-tail reads finite in-bounds data) lives in LDS.
// ---------------------------------------------------------------------------
__global__ __launch_bounds__(512) void erode21_wmma(const float* __restrict__ map,
                                                    _Float16* __restrict__ fe) {
  __shared__ _Float16 sH[88 * 88];            // binarized input halo (0/1)
  __shared__ unsigned short sB[21 * 32 * 32]; // B-operand table

  const int tid = threadIdx.x;
  const int bx = blockIdx.x, by = blockIdx.y, bz = blockIdx.z;
  const long long chbase =
      ((long long)(bz * NCH + EXPLORED_MAP)) * MDIM * MDIM;

  // Fill halo: g = (map != 0) ? 1 : 0, zero outside the image (matches the
  // zero-padded conv of (1 - frontier) in the reference).
  for (int idx = tid; idx < 88 * 88; idx += 512) {
    const int r  = idx / 88;
    const int cc = idx - r * 88;
    const int gr = by * 64 - 10 + r;
    const int gc = bx * 64 - 10 + cc;
    _Float16 v = (_Float16)0;
    if ((unsigned)gr < (unsigned)MDIM && (unsigned)gc < (unsigned)MDIM)
      v = (map[chbase + (long long)gr * MDIM + gc] != 0.0f) ? (_Float16)1
                                                            : (_Float16)0;
    sH[idx] = v;
  }
  // Stage the (block-invariant) B table into LDS.
  {
    const unsigned short* bt = g_btab.v;
    for (int idx = tid; idx < 21 * 32 * 32; idx += 512) sB[idx] = bt[idx];
  }
  __syncthreads();

  const int lane = tid & 31;
  const int wv   = tid >> 5;       // 0..15
  const int tx   = wv & 3;         // tile x within 64x64 region
  const int ty   = wv >> 2;        // tile y within 64x64 region
  const int m    = lane & 15;      // A-operand row
  const int hi   = (lane & 16) ? 16 : 0;  // A-operand K sub-base
  const int cb   = tx * 16;        // local column base of this tile

  v8f c = {};
  for (int ky = 0; ky < 21; ++ky) {
    const _Float16* rowp = &sH[(ty * 16 + m + ky) * 88 + cb];
    const _Float16* bp   = (const _Float16*)&sB[(ky * 32 + lane) * 32];

    // A1/B1: image columns K = 0..31 relative to (tile_x0 - 10)
    const v8h a_lo = *(const v8h*)(rowp + hi);
    const v8h a_hi = *(const v8h*)(rowp + hi + 8);
    v16h a1;
#pragma unroll
    for (int i = 0; i < 8; ++i) { a1[i] = a_lo[i]; a1[8 + i] = a_hi[i]; }

    const v8h b_lo = *(const v8h*)(bp);
    const v8h b_hi = *(const v8h*)(bp + 8);
    v16h b1;
#pragma unroll
    for (int i = 0; i < 8; ++i) { b1[i] = b_lo[i]; b1[8 + i] = b_hi[i]; }

    c = __builtin_amdgcn_wmma_f32_16x16x32_f16(false, a1, false, b1,
                                               (short)0, c, false, false);

    // Second K-chunk (input cols 32..35) only has nonzero weights for
    // ky in [3,17]; branch is wave-uniform so EXEC stays all-ones.
    // Both operands are full 16-element b128 loads; the tail elements are
    // multiplied by zero table entries (finite LDS data, no movs needed).
    if (ky >= 3 && ky <= 17) {
      const v8h a2_lo = *(const v8h*)(rowp + 32 + hi);
      const v8h a2_hi = *(const v8h*)(rowp + 40 + hi);
      v16h a2;
#pragma unroll
      for (int i = 0; i < 8; ++i) { a2[i] = a2_lo[i]; a2[8 + i] = a2_hi[i]; }

      const v8h b2_lo = *(const v8h*)(bp + 16);
      const v8h b2_hi = *(const v8h*)(bp + 24);
      v16h b2;
#pragma unroll
      for (int i = 0; i < 8; ++i) { b2[i] = b2_lo[i]; b2[8 + i] = b2_hi[i]; }

      c = __builtin_amdgcn_wmma_f32_16x16x32_f16(false, a2, false, b2,
                                                 (short)0, c, false, false);
    }
  }

  // Threshold: eroded frontier = 1 iff conv(1-frontier, disk21) == 0.
  // Sums are exact small integers, so < 0.5 test is exact.
  const int gx0  = bx * 64 + tx * 16;
  const int gy0  = by * 64 + ty * 16;
  const int n    = lane & 15;             // D column
  const int mrow = (lane & 16) ? 8 : 0;   // D row base per VGPR
  _Float16* feb = fe + (long long)bz * MDIM * MDIM;
#pragma unroll
  for (int v = 0; v < 8; ++v) {
    const _Float16 e = (c[v] < 0.5f) ? (_Float16)1 : (_Float16)0;
    feb[(gy0 + mrow + v) * MDIM + gx0 + n] = e;
  }
}

// ---------------------------------------------------------------------------
// Kernel 2: frontier border = clip(conv(fe, disk1), 0, 1) - fe.
// disk(1) is the 5-point cross; zero padding at the image border.
// ---------------------------------------------------------------------------
__global__ __launch_bounds__(256) void frontier_border(
    const _Float16* __restrict__ fe, float* __restrict__ out) {
  const int b = blockIdx.y;
  const int p = blockIdx.x * 256 + threadIdx.x;  // 0 .. MDIM*MDIM-1
  const int y = p >> 10;
  const int x = p & (MDIM - 1);
  const _Float16* f = fe + (long long)b * MDIM * MDIM;
  const float ctr = (float)f[p];
  const float up  = (y > 0)        ? (float)f[p - MDIM] : 0.0f;
  const float dn  = (y < MDIM - 1) ? (float)f[p + MDIM] : 0.0f;
  const float lf  = (x > 0)        ? (float)f[p - 1]    : 0.0f;
  const float rt  = (x < MDIM - 1) ? (float)f[p + 1]    : 0.0f;
  const float d = fminf(ctr + up + dn + lf + rt, 1.0f);
  out[(long long)b * MDIM * MDIM + p] = d - ctr;
}

extern "C" void kernel_launch(void* const* d_in, const int* in_sizes, int n_in,
                              void* d_out, int out_size, void* d_ws, size_t ws_size,
                              hipStream_t stream) {
  (void)in_sizes; (void)n_in; (void)out_size; (void)ws_size;
  const float* map = (const float*)d_in[0];   // (16, 8, 1024, 1024) fp32
  // d_in[1]/d_in[2] (disk kernels) are fixed disks; baked in at compile time.
  _Float16* fe = (_Float16*)d_ws;             // 16*1024*1024 f16 = 32 MB scratch
  float* out = (float*)d_out;                 // (16, 1, 1024, 1024) fp32

  dim3 g1(MDIM / 64, MDIM / 64, NBATCH);      // (16,16,16)
  erode21_wmma<<<g1, dim3(512), 0, stream>>>(map, fe);

  dim3 g2((MDIM * MDIM) / 256, NBATCH);
  frontier_border<<<g2, dim3(256), 0, stream>>>(fe, out);
}
